// MultiHeadAttention_10110353014975
// MI455X (gfx1250) — matrix operand order repaired and checked by reading, not compiled, not run on hardware
//
#include <hip/hip_runtime.h>
#include <hip/hip_bf16.h>

#define BB  2
#define TT  2048
#define DD  1024
#define HH  16
#define DKK 64

typedef __attribute__((ext_vector_type(16))) _Float16 v16h;
typedef __attribute__((ext_vector_type(8)))  _Float16 v8h;
typedef __attribute__((ext_vector_type(4)))  _Float16 v4h;
typedef __attribute__((ext_vector_type(8)))  float    v8f;

union Frag16 { v16h v; v8h h[2]; };

__device__ __forceinline__ v8f wmma_f16(v16h a, v16h b, v8f c) {
  // 8 args: (neg_a, A, neg_b, B, c_mod, C, reuse_a, reuse_b)
  return __builtin_amdgcn_wmma_f32_16x16x32_f16(false, a, false, b, (short)0, c,
                                                false, false);
}

// ---------------------------------------------------------------------------
// f32 -> f16 conversion (4 elements / thread)
// ---------------------------------------------------------------------------
__global__ void cvt_f32_to_f16(const float* __restrict__ src,
                               _Float16* __restrict__ dst, int n) {
  int i = (blockIdx.x * blockDim.x + threadIdx.x) * 4;
  if (i + 3 < n) {
    float4 f = *(const float4*)(src + i);
    v4h o;
    o[0] = (_Float16)f.x; o[1] = (_Float16)f.y;
    o[2] = (_Float16)f.z; o[3] = (_Float16)f.w;
    *(v4h*)(dst + i) = o;
  }
}

// ---------------------------------------------------------------------------
// C[M,N] = A[M,K] @ W[N,K]^T + bias   (torch Linear convention)
// MODE 0: store f32 row-major [M,N]
// MODE 1: store f16 [B,H,T,DK]   (Q / K projections, head-split)
// MODE 2: store f16 [B,H,DK,T]   (V projection, transposed for P.V WMMA)
// One wave owns a 16x64 C strip; 4 waves / block.
// ---------------------------------------------------------------------------
template <int MODE>
__global__ void __launch_bounds__(128)
gemm_wmma(const _Float16* __restrict__ A,
          const _Float16* __restrict__ W,
          const float* __restrict__ bias,
          void* __restrict__ out,
          int M, int N, int K) {
  const int lane = threadIdx.x & 31;
  const int wave = threadIdx.x >> 5;
  const int lm = lane & 15;   // column (B/C) or row (A) within tile
  const int lh = lane >> 4;   // half-wave select
  const int m0 = (blockIdx.y * 4 + wave) * 16;
  const int n0 = blockIdx.x * 64;

  const _Float16* Arow = A + (size_t)(m0 + lm) * K + lh * 16;

  v8f acc[4] = {};

  for (int k0 = 0; k0 < K; k0 += 32) {
    Frag16 a;
    a.h[0] = *(const v8h*)(Arow + k0);        // K = k0 + 16*lh + [0..7]
    a.h[1] = *(const v8h*)(Arow + k0 + 8);    // K = k0 + 16*lh + 8 + [0..7]
#pragma unroll
    for (int t = 0; t < 4; ++t) {
      const _Float16* Wrow = W + (size_t)(n0 + t * 16 + lm) * K + k0 + lh * 16;
      v16h bf = *(const v16h*)Wrow;           // 16 contiguous K halves
      acc[t] = wmma_f16(a.v, bf, acc[t]);
    }
  }

#pragma unroll
  for (int t = 0; t < 4; ++t) {
    const int n = n0 + t * 16 + lm;
    const float bz = bias[n];
#pragma unroll
    for (int i = 0; i < 8; ++i) {
      const int m = m0 + i + 8 * lh;          // C layout: M = i + 8*(lane/16)
      float val = acc[t][i] + bz;
      if (MODE == 0) {
        ((float*)out)[(size_t)m * N + n] = val;
      } else {
        const int b  = m / TT;
        const int tt = m % TT;
        const int h  = n / DKK;
        const int dk = n % DKK;
        _Float16* o16 = (_Float16*)out;
        if (MODE == 1)
          o16[(((size_t)b * HH + h) * TT + tt) * DKK + dk] = (_Float16)val;
        else
          o16[(((size_t)b * HH + h) * DKK + dk) * TT + tt] = (_Float16)val;
      }
    }
  }
}

// ---------------------------------------------------------------------------
// Flash attention: one wave per (b, h, 16-query tile); 64 keys per iteration.
// Qb,Kb: [B,H,T,DK] f16 ; Vt: [B,H,DK,T] f16 ; mask: [B,T,T] i32
// Ob: [B,T,D] f16 (feeds the Wo GEMM)
// launch_bounds(128,1): keep Q frags + 4 score tiles + 4 O tiles resident
// (no scratch spills). Row sums via WMMA against ones (rowsum(P) = P @ 1).
// ---------------------------------------------------------------------------
__global__ void __launch_bounds__(128, 1)
attn_wmma(const _Float16* __restrict__ Qb,
          const _Float16* __restrict__ Kb,
          const _Float16* __restrict__ Vt,
          const int* __restrict__ mask,
          _Float16* __restrict__ Ob) {
  __shared__ _Float16 Plds[4][16 * 64];  // per-wave P staging (C->A relayout)

  const int lane = threadIdx.x & 31;
  const int wave = threadIdx.x >> 5;
  const int lm = lane & 15;
  const int lh = lane >> 4;

  const int gid = blockIdx.x * 4 + wave;
  const int b  = gid / (HH * (TT / 16));
  const int r  = gid % (HH * (TT / 16));
  const int h  = r / (TT / 16);
  const int q0 = (r % (TT / 16)) * 16;

  const size_t bh = (size_t)b * HH + h;

  // Q tile 16x64 as two A-fragments (dk 0..31, dk 32..63)
  const _Float16* Qrow = Qb + (bh * TT + q0 + lm) * DKK;
  Frag16 a0, a1;
  a0.h[0] = *(const v8h*)(Qrow + lh * 16);
  a0.h[1] = *(const v8h*)(Qrow + lh * 16 + 8);
  a1.h[0] = *(const v8h*)(Qrow + 32 + lh * 16);
  a1.h[1] = *(const v8h*)(Qrow + 32 + lh * 16 + 8);

  v16h ones;
#pragma unroll
  for (int j = 0; j < 16; ++j) ones[j] = (_Float16)1.0f;

  float mst[8], lst[8];
#pragma unroll
  for (int i = 0; i < 8; ++i) { mst[i] = -3.0e38f; lst[i] = 0.0f; }
  v8f o[4] = {};

  const float scale = 0.125f;  // 1/sqrt(64)
  const int* maskq = mask + ((size_t)b * TT + q0) * TT;  // rows q0..q0+15

  for (int j0 = 0; j0 < TT; j0 += 64) {
    if (j0 + 64 < TT) {  // prefetch next key/value blocks -> global_prefetch_b8
      __builtin_prefetch(Kb + (bh * TT + j0 + 64 + lm) * DKK, 0, 1);
      __builtin_prefetch(Vt + (bh * DKK + lm) * TT + j0 + 64, 0, 1);
    }

    // ---- S = Q @ K^T : four 16x16 score tiles covering 64 keys ----
    v8f s[4] = {};
#pragma unroll
    for (int half = 0; half < 4; ++half) {
      const _Float16* Krow = Kb + (bh * TT + j0 + half * 16 + lm) * DKK;
      v16h k0f = *(const v16h*)(Krow + lh * 16);        // dk 0..31 slice
      v16h k1f = *(const v16h*)(Krow + 32 + lh * 16);   // dk 32..63 slice
      s[half] = wmma_f16(a0.v, k0f, s[half]);
      s[half] = wmma_f16(a1.v, k1f, s[half]);
    }

    // ---- mask: batch all 32 loads so they clause under one wait ----
    int mv[4][8];
#pragma unroll
    for (int half = 0; half < 4; ++half)
#pragma unroll
      for (int i = 0; i < 8; ++i)
        mv[half][i] = maskq[(size_t)(i + 8 * lh) * TT + j0 + half * 16 + lm];

    // scale + mask in place (no extra 32-register array)
#pragma unroll
    for (int half = 0; half < 4; ++half)
#pragma unroll
      for (int i = 0; i < 8; ++i) {
        const float x = s[half][i] * scale;
        s[half][i] = (mv[half][i] == 0) ? -1.0e9f : x;
      }

    // ---- row max: local reduce, then batched 16-lane butterflies ----
    float rm[8];
#pragma unroll
    for (int i = 0; i < 8; ++i)
      rm[i] = fmaxf(fmaxf(s[0][i], s[1][i]), fmaxf(s[2][i], s[3][i]));
#pragma unroll
    for (int off = 1; off <= 8; off <<= 1)
#pragma unroll
      for (int i = 0; i < 8; ++i)   // 8 independent bpermutes in flight
        rm[i] = fmaxf(rm[i], __shfl_xor(rm[i], off));

    float alpha[8];
#pragma unroll
    for (int i = 0; i < 8; ++i) {
      const float mnew = fmaxf(mst[i], rm[i]);
      alpha[i] = __expf(mst[i] - mnew);
      mst[i] = mnew;
#pragma unroll
      for (int t = 0; t < 4; ++t) o[t][i] *= alpha[i];
    }

    // ---- P = exp(S - m), staged to LDS row-major 16x64 (f16) ----
#pragma unroll
    for (int half = 0; half < 4; ++half)
#pragma unroll
      for (int i = 0; i < 8; ++i) {
        const float p = __expf(s[half][i] - mst[i]);
        Plds[wave][(i + 8 * lh) * 64 + half * 16 + lm] = (_Float16)p;
      }

    __builtin_amdgcn_wave_barrier();  // fence compiler over same-wave LDS RAW

    Frag16 p0, p1;  // A-fragments: keys 0..31 and 32..63 of this block
    p0.h[0] = *(const v8h*)&Plds[wave][lm * 64 + lh * 16];
    p0.h[1] = *(const v8h*)&Plds[wave][lm * 64 + lh * 16 + 8];
    p1.h[0] = *(const v8h*)&Plds[wave][lm * 64 + 32 + lh * 16];
    p1.h[1] = *(const v8h*)&Plds[wave][lm * 64 + 32 + lh * 16 + 8];

    // ---- row sums via WMMA against ones: every column = rowsum(P) ----
    v8f lsum = {};
    lsum = wmma_f16(p0.v, ones, lsum);
    lsum = wmma_f16(p1.v, ones, lsum);
#pragma unroll
    for (int i = 0; i < 8; ++i) lst[i] = lst[i] * alpha[i] + lsum[i];

    // ---- O += P @ V : contraction over 64 keys, dk in 4 N-tiles ----
#pragma unroll
    for (int t = 0; t < 4; ++t) {
      const _Float16* Vrow = Vt + (bh * DKK + t * 16 + lm) * TT + j0 + lh * 16;
      v16h vf0 = *(const v16h*)Vrow;         // keys j0+0..31 (V transposed)
      v16h vf1 = *(const v16h*)(Vrow + 32);  // keys j0+32..63
      o[t] = wmma_f16(p0.v, vf0, o[t]);
      o[t] = wmma_f16(p1.v, vf1, o[t]);
    }
  }

  // ---- normalize and store [B,T,D] f16 ----
#pragma unroll
  for (int i = 0; i < 8; ++i) {
    const int row = q0 + i + 8 * lh;
    const float inv = 1.0f / lst[i];
#pragma unroll
    for (int t = 0; t < 4; ++t) {
      const int col = h * DKK + t * 16 + lm;
      Ob[((size_t)b * TT + row) * DD + col] = (_Float16)(o[t][i] * inv);
    }
  }
}

// ---------------------------------------------------------------------------
extern "C" void kernel_launch(void* const* d_in, const int* in_sizes, int n_in,
                              void* d_out, int out_size, void* d_ws, size_t ws_size,
                              hipStream_t stream) {
  (void)in_sizes; (void)n_in; (void)out_size; (void)ws_size;

  const float* q    = (const float*)d_in[0];
  const float* k    = (const float*)d_in[1];
  const float* v    = (const float*)d_in[2];
  const int*   mask = (const int*)d_in[3];
  const float* Wq   = (const float*)d_in[4];
  const float* bq   = (const float*)d_in[5];
  const float* Wk   = (const float*)d_in[6];
  const float* bk   = (const float*)d_in[7];
  const float* Wv   = (const float*)d_in[8];
  const float* bv   = (const float*)d_in[9];
  const float* Wo   = (const float*)d_in[10];
  const float* bo   = (const float*)d_in[11];
  float* out = (float*)d_out;

  const size_t MT = (size_t)BB * TT;   // 4096 rows
  const size_t ND = MT * DD;           // 4M activations
  const size_t NW = (size_t)DD * DD;   // 1M weights

  _Float16* base = (_Float16*)d_ws;    // 64 MB of f16 scratch total
  _Float16* hq  = base;
  _Float16* hk  = hq  + ND;
  _Float16* hv  = hk  + ND;
  _Float16* hWq = hv  + ND;
  _Float16* hWk = hWq + NW;
  _Float16* hWv = hWk + NW;
  _Float16* hWo = hWv + NW;
  _Float16* Qb  = hWo + NW;   // [B,H,T,DK]
  _Float16* Kb  = Qb  + ND;   // [B,H,T,DK]
  _Float16* Vt  = Kb  + ND;   // [B,H,DK,T]
  _Float16* AO  = Vt  + ND;   // [B,T,D]

  dim3 cblk(256);
  auto cvt = [&](const float* s, _Float16* d, size_t n) {
    dim3 g((unsigned)((n / 4 + 255) / 256));
    cvt_f32_to_f16<<<g, cblk, 0, stream>>>(s, d, (int)n);
  };
  cvt(q, hq, ND);  cvt(k, hk, ND);  cvt(v, hv, ND);
  cvt(Wq, hWq, NW); cvt(Wk, hWk, NW); cvt(Wv, hWv, NW); cvt(Wo, hWo, NW);

  dim3 gg(DD / 64, (unsigned)(MT / 64));  // (16, 64)
  dim3 gb(128);                           // 4 waves
  gemm_wmma<1><<<gg, gb, 0, stream>>>(hq, hWq, bq, Qb, (int)MT, DD, DD);
  gemm_wmma<1><<<gg, gb, 0, stream>>>(hk, hWk, bk, Kb, (int)MT, DD, DD);
  gemm_wmma<2><<<gg, gb, 0, stream>>>(hv, hWv, bv, Vt, (int)MT, DD, DD);

  dim3 ag((unsigned)(BB * HH * (TT / 16) / 4));  // 1024 blocks, 1 wave/(b,h,qtile)
  attn_wmma<<<ag, gb, 0, stream>>>(Qb, Kb, Vt, mask, AO);

  gemm_wmma<0><<<gg, gb, 0, stream>>>(AO, hWo, bo, out, (int)MT, DD, DD);
}
